// Attention_90752658964988
// MI455X (gfx1250) — matrix operand order repaired and checked by reading, not compiled, not run on hardware
//
#include <hip/hip_runtime.h>
#include <hip/hip_bf16.h>

// Problem constants (reference: B=32, N=1024, D=512, H=8)
#define B_ 32
#define N_ 1024
#define D_ 512
#define H_ 8

typedef __attribute__((ext_vector_type(16))) __bf16 v16bf;
typedef __attribute__((ext_vector_type(8)))  __bf16 v8bf;
typedef __attribute__((ext_vector_type(8)))  float  v8f;

union FragU { v16bf v; v8bf h[2]; };

__device__ __forceinline__ int laneid() { return threadIdx.x & 31; }

// A-fragment (16x32 bf16, M x K), source row-major [16][ld] starting at k0.
// ISA layout: lanes 0-15 hold M=lane, K={0..7,16..23}; lanes 16-31 K={8..15,24..31}.
__device__ __forceinline__ v16bf load_frag_a(const __bf16* base, int ld, int k0) {
  const int lane = laneid();
  const __bf16* p = base + (size_t)(lane & 15) * ld + k0 + ((lane >> 4) << 3);
  FragU f;
  f.h[0] = *(const v8bf*)(p);
  f.h[1] = *(const v8bf*)(p + 16);
  return f.v;
}

// B-fragment (32x16 bf16, K x N). Source stored N-major: row n holds contiguous K.
// Lanes 0-15: N=lane, K={0..7,16..23}; lanes 16-31: N=lane-16, K={8..15,24..31}.
__device__ __forceinline__ v16bf load_frag_b(const __bf16* base, int ld, int k0) {
  const int lane = laneid();
  const __bf16* p = base + (size_t)(lane & 15) * ld + k0 + ((lane >> 4) << 3);
  FragU f;
  f.h[0] = *(const v8bf*)(p);
  f.h[1] = *(const v8bf*)(p + 16);
  return f.v;
}

__device__ __forceinline__ v8f wmma_bf16(v16bf a, v16bf b, v8f c) {
  return __builtin_amdgcn_wmma_f32_16x16x32_bf16(false, a, false, b, (short)0, c,
                                                 false, false);
}

__device__ __forceinline__ float wave_max(float v) {
  #pragma unroll
  for (int o = 16; o > 0; o >>= 1) v = fmaxf(v, __shfl_xor(v, o, 32));
  return v;
}
__device__ __forceinline__ float wave_sum(float v) {
  #pragma unroll
  for (int o = 16; o > 0; o >>= 1) v += __shfl_xor(v, o, 32);
  return v;
}

// ---------------- conversion kernels ----------------
__global__ void cvt_bf16_kernel(const float* __restrict__ src,
                                __bf16* __restrict__ dst, int n) {
  int i = blockIdx.x * blockDim.x + threadIdx.x;
  int stride = gridDim.x * blockDim.x;
  for (; i < n; i += stride) dst[i] = (__bf16)src[i];
}

// Wo [D, H*D] (dout, j=d*H+h)  ->  Wo2 [H][dout][d] bf16 (B-fragment friendly)
__global__ void reorder_wo_kernel(const float* __restrict__ wo,
                                  __bf16* __restrict__ wo2) {
  int i = blockIdx.x * blockDim.x + threadIdx.x;
  int stride = gridDim.x * blockDim.x;
  const int total = H_ * D_ * D_;
  for (; i < total; i += stride) {
    int h = i / (D_ * D_);
    int r = i % (D_ * D_);
    int dout = r / D_;
    int d = r % D_;
    wo2[i] = (__bf16)wo[(size_t)dout * (H_ * D_) + (size_t)d * H_ + h];
  }
}

// ---------------- per-head projection GEMM (64x64 macro-tile per wave) ------
// Y[b,h,n,e] = sum_d X[b,n,d] * W[h,e,d] + bias[h,e]
// 4x4 register tiling: 16 accumulators, A-frags reused 4x, B-frags 4x.
// transposed==0: store [B,H,N,D]; transposed==1: store [B,H,D,N] (for V).
__global__ __launch_bounds__(256) void proj_kernel(
    const __bf16* __restrict__ X,   // [B*N, D] bf16
    const __bf16* __restrict__ W,   // [H, D, D] bf16 (e-major, d contiguous)
    const float* __restrict__ bias, // [H, D] fp32
    __bf16* __restrict__ out, int transposed) {
  const int wave = threadIdx.x >> 5;
  const int lane = threadIdx.x & 31;
  const int h = blockIdx.y;
  const int tile = blockIdx.x * (blockDim.x >> 5) + wave;
  const int mcTiles = D_ / 64;                  // 8 macro cols
  const int mr = tile / mcTiles;                // over (B*N)/64
  const int mc = tile % mcTiles;

  const __bf16* abase = X + (size_t)mr * 64 * D_;
  const __bf16* bbase = W + ((size_t)h * D_ + (size_t)mc * 64) * D_;

  v8f acc[4][4];
  #pragma unroll
  for (int j = 0; j < 4; ++j) {
    const float bval = bias[h * D_ + mc * 64 + j * 16 + (lane & 15)];
    #pragma unroll
    for (int i = 0; i < 4; ++i)
      #pragma unroll
      for (int r = 0; r < 8; ++r) acc[i][j][r] = bval;
  }

  #pragma unroll 2
  for (int k0 = 0; k0 < D_; k0 += 32) {
    v16bf a[4];
    #pragma unroll
    for (int i = 0; i < 4; ++i)
      a[i] = load_frag_a(abase + (size_t)i * 16 * D_, D_, k0);
    #pragma unroll
    for (int j = 0; j < 4; ++j) {
      v16bf b = load_frag_b(bbase + (size_t)j * 16 * D_, D_, k0);
      #pragma unroll
      for (int i = 0; i < 4; ++i) acc[i][j] = wmma_bf16(a[i], b, acc[i][j]);
    }
  }

  #pragma unroll
  for (int i = 0; i < 4; ++i) {
    #pragma unroll
    for (int j = 0; j < 4; ++j) {
      const int col = mc * 64 + j * 16 + (lane & 15);
      #pragma unroll
      for (int r = 0; r < 8; ++r) {
        const int g = mr * 64 + i * 16 + r + ((lane >> 4) << 3);  // row in [B*N)
        const int bb = g >> 10;                                   // / N_
        const int nn = g & (N_ - 1);
        size_t idx;
        if (!transposed)
          idx = (((size_t)bb * H_ + h) * N_ + nn) * D_ + col;
        else
          idx = (((size_t)bb * H_ + h) * D_ + col) * N_ + nn;
        out[idx] = (__bf16)acc[i][j][r];
      }
    }
  }
}

// ---------------- fused attention: scores -> softmax -> P @ V ----------------
// One block (8 waves) handles one (b,h) and a 16-row query strip.
// LDS: S fp32 [16][N] (64KB) + P bf16 [16][N] (32KB).
__global__ __launch_bounds__(256) void attn_kernel(
    const __bf16* __restrict__ qh,   // [B,H,N,D]
    const __bf16* __restrict__ kh,   // [B,H,N,D]
    const __bf16* __restrict__ vhT,  // [B,H,D,N]
    __bf16* __restrict__ attn) {     // [B,H,N,D]
  extern __shared__ char smem[];
  float*  S = (float*)smem;                          // [16][N_]
  __bf16* P = (__bf16*)(smem + (size_t)16 * N_ * 4); // [16][N_]

  const int wave = threadIdx.x >> 5;
  const int lane = threadIdx.x & 31;
  const int qTile = blockIdx.x & (N_ / 16 - 1);      // 64 strips per (b,h)
  const int bh = blockIdx.x >> 6;
  const size_t head = (size_t)bh * N_ * D_;

  const __bf16* qbase = qh + head + (size_t)qTile * 16 * D_;
  const float scale = 0.044194173824159216f;         // 1/sqrt(512)

  // ---- pass 1: S[16][1024] = Qtile @ K^T ; wave owns 8 key col-tiles ----
  v8f acc[8];
  #pragma unroll
  for (int t = 0; t < 8; ++t)
    #pragma unroll
    for (int r = 0; r < 8; ++r) acc[t][r] = 0.f;

  for (int k0 = 0; k0 < D_; k0 += 32) {
    v16bf a = load_frag_a(qbase, D_, k0);
    #pragma unroll
    for (int t = 0; t < 8; ++t) {
      const __bf16* bbase = kh + head + (size_t)(wave * 8 + t) * 16 * D_;
      v16bf b = load_frag_b(bbase, D_, k0);
      acc[t] = wmma_bf16(a, b, acc[t]);
    }
  }
  #pragma unroll
  for (int t = 0; t < 8; ++t) {
    const int col = (wave * 8 + t) * 16 + (lane & 15);
    #pragma unroll
    for (int r = 0; r < 8; ++r) {
      const int row = r + ((lane >> 4) << 3);
      S[row * N_ + col] = acc[t][r] * scale;
    }
  }
  __syncthreads();

  // ---- pass 2: softmax; each wave handles 2 of the 16 rows ----
  #pragma unroll
  for (int rr = 0; rr < 2; ++rr) {
    const int row = wave * 2 + rr;
    float* Sr = S + row * N_;
    float m = -1e30f;
    for (int c = lane; c < N_; c += 32) m = fmaxf(m, Sr[c]);
    m = wave_max(m);
    float sum = 0.f;
    for (int c = lane; c < N_; c += 32) {
      float e = __expf(Sr[c] - m);
      Sr[c] = e;
      sum += e;
    }
    sum = wave_sum(sum);
    const float inv = 1.f / sum;
    __bf16* Pr = P + row * N_;
    for (int c = lane; c < N_; c += 32) Pr[c] = (__bf16)(Sr[c] * inv);
  }
  __syncthreads();

  // ---- pass 3: O = P @ V ; wave owns 4 col-tiles of D (64 cols) ----
  v8f o[4];
  #pragma unroll
  for (int t = 0; t < 4; ++t)
    #pragma unroll
    for (int r = 0; r < 8; ++r) o[t][r] = 0.f;

  for (int k0 = 0; k0 < N_; k0 += 32) {
    v16bf a = load_frag_a(P, N_, k0);   // A from LDS (ds path)
    #pragma unroll
    for (int t = 0; t < 4; ++t) {
      const __bf16* bbase = vhT + head + (size_t)(wave * 4 + t) * 16 * N_;
      v16bf b = load_frag_b(bbase, N_, k0);
      o[t] = wmma_bf16(a, b, o[t]);
    }
  }
  #pragma unroll
  for (int t = 0; t < 4; ++t) {
    const int col = (wave * 4 + t) * 16 + (lane & 15);
    #pragma unroll
    for (int r = 0; r < 8; ++r) {
      const int row = qTile * 16 + r + ((lane >> 4) << 3);
      attn[head + (size_t)row * D_ + col] = (__bf16)o[t][r];
    }
  }
}

// ---------------- output projection (64x64 macro-tile per wave) --------------
// rep[b,n,dout] = bo[dout] + sum_h sum_d attn[b,h,n,d] * Wo2[h][dout][d]
__global__ __launch_bounds__(256) void outproj_kernel(
    const __bf16* __restrict__ attn, // [B,H,N,D]
    const __bf16* __restrict__ wo2,  // [H][D(out)][D(d)]
    const float* __restrict__ bo,    // [D]
    float* __restrict__ out) {       // [B,N,D] fp32
  const int wave = threadIdx.x >> 5;
  const int lane = threadIdx.x & 31;
  const int tile = blockIdx.x * (blockDim.x >> 5) + wave;
  const int mcTiles = D_ / 64;                  // 8
  const int mr = tile / mcTiles;                // over (B*N)/64
  const int mc = tile % mcTiles;
  const int g0 = mr * 64;
  const int bb = g0 >> 10;
  const int n0 = g0 & (N_ - 1);

  v8f acc[4][4];
  #pragma unroll
  for (int j = 0; j < 4; ++j) {
    const float bval = bo[mc * 64 + j * 16 + (lane & 15)];
    #pragma unroll
    for (int i = 0; i < 4; ++i)
      #pragma unroll
      for (int r = 0; r < 8; ++r) acc[i][j][r] = bval;
  }

  for (int h = 0; h < H_; ++h) {
    const __bf16* abase = attn + (((size_t)bb * H_ + h) * N_ + n0) * D_;
    const __bf16* bbase = wo2 + ((size_t)h * D_ + (size_t)mc * 64) * D_;
    #pragma unroll 2
    for (int k0 = 0; k0 < D_; k0 += 32) {
      v16bf a[4];
      #pragma unroll
      for (int i = 0; i < 4; ++i)
        a[i] = load_frag_a(abase + (size_t)i * 16 * D_, D_, k0);
      #pragma unroll
      for (int j = 0; j < 4; ++j) {
        v16bf b = load_frag_b(bbase + (size_t)j * 16 * D_, D_, k0);
        #pragma unroll
        for (int i = 0; i < 4; ++i) acc[i][j] = wmma_bf16(a[i], b, acc[i][j]);
      }
    }
  }

  #pragma unroll
  for (int i = 0; i < 4; ++i) {
    #pragma unroll
    for (int j = 0; j < 4; ++j) {
      const int col = mc * 64 + j * 16 + (lane & 15);
      #pragma unroll
      for (int r = 0; r < 8; ++r) {
        const int g = g0 + i * 16 + r + ((lane >> 4) << 3);
        out[(size_t)g * D_ + col] = acc[i][j][r];
      }
    }
  }
}

// ---------------- host launch ----------------
extern "C" void kernel_launch(void* const* d_in, const int* in_sizes, int n_in,
                              void* d_out, int out_size, void* d_ws, size_t ws_size,
                              hipStream_t stream) {
  const float* k  = (const float*)d_in[0];
  const float* v  = (const float*)d_in[1];
  const float* q  = (const float*)d_in[2];
  const float* Wk = (const float*)d_in[3];
  const float* bk = (const float*)d_in[4];
  const float* Wv = (const float*)d_in[5];
  const float* bv = (const float*)d_in[6];
  const float* Wq = (const float*)d_in[7];
  const float* bq = (const float*)d_in[8];
  const float* Wo = (const float*)d_in[9];
  const float* bo = (const float*)d_in[10];
  float* out = (float*)d_out;

  const int nX = B_ * N_ * D_;                     // 16,777,216
  const int nW = H_ * D_ * D_;                     // 2,097,152
  const size_t nHead = (size_t)B_ * H_ * N_ * D_;  // 134,217,728

  char* ws = (char*)d_ws;
  size_t off = 0;
  auto alloc = [&](size_t bytes) -> char* {
    char* p = ws + off;
    off += (bytes + 255) & ~(size_t)255;
    return p;
  };
  __bf16* qb   = (__bf16*)alloc((size_t)nX * 2);
  __bf16* kb   = (__bf16*)alloc((size_t)nX * 2);
  __bf16* vb   = (__bf16*)alloc((size_t)nX * 2);
  __bf16* Wqb  = (__bf16*)alloc((size_t)nW * 2);
  __bf16* Wkb  = (__bf16*)alloc((size_t)nW * 2);
  __bf16* Wvb  = (__bf16*)alloc((size_t)nW * 2);
  __bf16* Wo2b = (__bf16*)alloc((size_t)nW * 2);
  __bf16* qhB  = (__bf16*)alloc(nHead * 2);
  __bf16* khB  = (__bf16*)alloc(nHead * 2);
  __bf16* vhTB = (__bf16*)alloc(nHead * 2);
  __bf16* attB = (__bf16*)alloc(nHead * 2);

  // 1) convert inputs/weights to bf16; reorder Wo
  cvt_bf16_kernel<<<4096, 256, 0, stream>>>(q, qb, nX);
  cvt_bf16_kernel<<<4096, 256, 0, stream>>>(k, kb, nX);
  cvt_bf16_kernel<<<4096, 256, 0, stream>>>(v, vb, nX);
  cvt_bf16_kernel<<<2048, 256, 0, stream>>>(Wq, Wqb, nW);
  cvt_bf16_kernel<<<2048, 256, 0, stream>>>(Wk, Wkb, nW);
  cvt_bf16_kernel<<<2048, 256, 0, stream>>>(Wv, Wvb, nW);
  reorder_wo_kernel<<<2048, 256, 0, stream>>>(Wo, Wo2b);

  // 2) per-head projections: (32768/64)*(512/64)=4096 macro tiles/head,
  //    8 waves/block -> 512 blocks per head
  dim3 pgrid(512, H_, 1);
  proj_kernel<<<pgrid, 256, 0, stream>>>(qb, Wqb, bq, qhB, 0);
  proj_kernel<<<pgrid, 256, 0, stream>>>(kb, Wkb, bk, khB, 0);
  proj_kernel<<<pgrid, 256, 0, stream>>>(vb, Wvb, bv, vhTB, 1);  // V transposed

  // 3) fused attention: one block per (b,h,16 query rows); 96KB LDS
  const int shmem = 16 * N_ * 4 + 16 * N_ * 2;  // 98304 bytes
  (void)hipFuncSetAttribute((const void*)attn_kernel,
                            hipFuncAttributeMaxDynamicSharedMemorySize, shmem);
  attn_kernel<<<B_ * H_ * (N_ / 16), 256, shmem, stream>>>(qhB, khB, vhTB, attB);

  // 4) output projection (K = H*D = 4096), 512 blocks of 8 waves
  outproj_kernel<<<512, 256, 0, stream>>>(attB, Wo2b, bo, out);
}
